// Pointnet2_center_47863115546832
// MI455X (gfx1250) — matrix operand order repaired and checked by reading, not compiled, not run on hardware
//
#include <hip/hip_runtime.h>
#include <cstdint>

typedef _Float16 v16h __attribute__((ext_vector_type(16)));
typedef _Float16 v8h  __attribute__((ext_vector_type(8)));
typedef _Float16 v4h  __attribute__((ext_vector_type(4)));
typedef float    v8f  __attribute__((ext_vector_type(8)));

#define EPS_BN 1e-5f

// ---------------------------------------------------------------------------
// Async global->LDS staging (CDNA5 ASYNCcnt path) with VGPR-staging fallback.
// Builtin signature (from clang diagnostics):
//   void __builtin_amdgcn_global_load_async_to_lds_b128(v4i AS1* src, v4i AS3* dst,
//                                                       imm offset, imm cpol)
// ---------------------------------------------------------------------------
#if __has_builtin(__builtin_amdgcn_global_load_async_to_lds_b128) && \
    __has_builtin(__builtin_amdgcn_global_load_async_to_lds_b64)
#define ASYNC_LDS 1
#else
#define ASYNC_LDS 0
#endif

#if ASYNC_LDS
#define AS_GLOBAL __attribute__((address_space(1)))
#define AS_LOCAL  __attribute__((address_space(3)))
typedef int ai4 __attribute__((vector_size(4 * sizeof(int))));
typedef int ai2 __attribute__((vector_size(2 * sizeof(int))));
__device__ __forceinline__ void async_b128(const _Float16* g, _Float16* l) {
  __builtin_amdgcn_global_load_async_to_lds_b128((AS_GLOBAL ai4*)g, (AS_LOCAL ai4*)l, 0, 0);
}
__device__ __forceinline__ void async_b64(const _Float16* g, _Float16* l) {
  __builtin_amdgcn_global_load_async_to_lds_b64((AS_GLOBAL ai2*)g, (AS_LOCAL ai2*)l, 0, 0);
}
__device__ __forceinline__ void wait_async() {
#if __has_builtin(__builtin_amdgcn_s_wait_asynccnt)
  __builtin_amdgcn_s_wait_asynccnt(0);
#else
  asm volatile("s_wait_asynccnt 0x0" ::: "memory");
#endif
}
#endif

// ---------------------------------------------------------------------------
// Furthest point sampling: one workgroup per batch, mind[] in LDS, xyz from L2.
// ---------------------------------------------------------------------------
__global__ __launch_bounds__(256) void fps_kernel(const float* __restrict__ xyz,
                                                  int* __restrict__ idx,
                                                  int N, int npoint) {
  __shared__ float mind[8192];
  __shared__ float redv[256];
  __shared__ int   redi[256];
  __shared__ float curp[3];
  const int b = blockIdx.x;
  const int t = threadIdx.x;
  const float* base = xyz + (size_t)b * N * 3;
  if (t == 0) {
    idx[b * npoint] = 0;
    curp[0] = base[0]; curp[1] = base[1]; curp[2] = base[2];
  }
  __syncthreads();
  for (int it = 1; it < npoint; ++it) {
    const float cx = curp[0], cy = curp[1], cz = curp[2];
    float bv = -1.0f; int bi = 0x7fffffff;
    for (int i = t; i < N; i += 256) {
      float dx = base[i * 3 + 0] - cx;
      float dy = base[i * 3 + 1] - cy;
      float dz = base[i * 3 + 2] - cz;
      float d2 = dx * dx + dy * dy + dz * dz;
      float mv = (it == 1) ? d2 : fminf(mind[i], d2);
      mind[i] = mv;
      if (mv > bv) { bv = mv; bi = i; }
    }
    redv[t] = bv; redi[t] = bi;
    __syncthreads();
    for (int s = 128; s > 0; s >>= 1) {
      if (t < s) {
        float ov = redv[t + s]; int oi = redi[t + s];
        if (ov > redv[t] || (ov == redv[t] && oi < redi[t])) { redv[t] = ov; redi[t] = oi; }
      }
      __syncthreads();
    }
    if (t == 0) {
      int n = redi[0];
      idx[b * npoint + it] = n;
      curp[0] = base[n * 3 + 0]; curp[1] = base[n * 3 + 1]; curp[2] = base[n * 3 + 2];
    }
    __syncthreads();
  }
}

// ---------------------------------------------------------------------------
// Gather sampled centers: new_xyz[b,i] = xyz[b, idx[b,i]]
// ---------------------------------------------------------------------------
__global__ void gather_xyz_kernel(const float* __restrict__ xyz, const int* __restrict__ idx,
                                  float* __restrict__ out, int N, int np, int total) {
  int t = blockIdx.x * blockDim.x + threadIdx.x;
  if (t >= total) return;
  int b = t / np;
  int n = idx[t];
  const float* src = xyz + ((size_t)b * N + n) * 3;
  out[t * 3 + 0] = src[0];
  out[t * 3 + 1] = src[1];
  out[t * 3 + 2] = src[2];
}

// ---------------------------------------------------------------------------
// Ball query: first nsample in-index-order within radius, padded with first.
// One wave32 per center; ballot + prefix popcount.
// ---------------------------------------------------------------------------
__global__ __launch_bounds__(256) void ballquery_kernel(const float* __restrict__ xyz,
                                                        const float* __restrict__ new_xyz,
                                                        int* __restrict__ nidx,
                                                        int B, int N, int np, int ns, float r2) {
  int gw = (int)(((size_t)blockIdx.x * blockDim.x + threadIdx.x) >> 5);
  int lane = threadIdx.x & 31;
  if (gw >= B * np) return;
  int b = gw / np;
  const float* ctr = new_xyz + (size_t)gw * 3;
  const float cx = ctr[0], cy = ctr[1], cz = ctr[2];
  const float* xb = xyz + (size_t)b * N * 3;
  int* out = nidx + (size_t)gw * ns;
  int cnt = 0;
  int first = -1;
  for (int basei = 0; basei < N && cnt < ns; basei += 32) {
    int j = basei + lane;
    bool pred = false;
    if (j < N) {
      float dx = xb[j * 3 + 0] - cx;
      float dy = xb[j * 3 + 1] - cy;
      float dz = xb[j * 3 + 2] - cz;
      pred = (dx * dx + dy * dy + dz * dz) <= r2;
    }
    unsigned mask = (unsigned)__ballot(pred);
    if (first < 0 && mask) first = basei + __builtin_ctz(mask);
    if (pred) {
      int pos = cnt + __popc(mask & ((1u << lane) - 1u));
      if (pos < ns) out[pos] = j;
    }
    cnt += __popc(mask);
  }
  if (cnt < ns) {
    if (first < 0) first = 0;
    for (int p = cnt + lane; p < ns; p += 32) out[p] = first;
  }
}

// ---------------------------------------------------------------------------
// Assemble grouped features X[row, Kpad] (f16): [rel_xyz(3) | feats(C) | 0-pad]
// ---------------------------------------------------------------------------
__global__ void group_kernel(const float* __restrict__ xyz, const float* __restrict__ new_xyz,
                             const _Float16* __restrict__ feats, const int* __restrict__ nidx,
                             _Float16* __restrict__ Xout,
                             int N, int np, int ns, int C, int Kpad, int total) {
  int row = blockIdx.x * blockDim.x + threadIdx.x;
  if (row >= total) return;
  int pi = (row / ns) % np;
  int b  = row / (ns * np);
  int n = nidx[row];
  const float* p = xyz + ((size_t)b * N + n) * 3;
  const float* q = new_xyz + ((size_t)b * np + pi) * 3;
  _Float16* o = Xout + (size_t)row * Kpad;
  o[0] = (_Float16)(p[0] - q[0]);
  o[1] = (_Float16)(p[1] - q[1]);
  o[2] = (_Float16)(p[2] - q[2]);
  int c = 3;
  if (C > 0) {
    const _Float16* f = feats + ((size_t)b * N + n) * C;
    for (int k = 0; k < C; ++k) o[3 + k] = f[k];
    c = 3 + C;
  }
  for (; c < Kpad; ++c) o[c] = (_Float16)0.f;
}

// ---------------------------------------------------------------------------
// Stage-4 pack: rows = b*32+j, features = [xyz3(3) | f3(512) | 0-pad]
// ---------------------------------------------------------------------------
__global__ void pack4_kernel(const float* __restrict__ xyz3, const _Float16* __restrict__ f3,
                             _Float16* __restrict__ Xout, int C, int Kpad, int total) {
  int row = blockIdx.x * blockDim.x + threadIdx.x;
  if (row >= total) return;
  _Float16* o = Xout + (size_t)row * Kpad;
  const float* p = xyz3 + (size_t)row * 3;
  o[0] = (_Float16)p[0]; o[1] = (_Float16)p[1]; o[2] = (_Float16)p[2];
  const _Float16* f = f3 + (size_t)row * C;
  for (int k = 0; k < C; ++k) o[3 + k] = f[k];
  for (int c = 3 + C; c < Kpad; ++c) o[c] = (_Float16)0.f;
}

// ---------------------------------------------------------------------------
// Per-layer prep: Wh[N,Kpad] = f16(W[N,K]) zero-padded; ss = {scale[N], shift[N]}
// ---------------------------------------------------------------------------
__global__ void prep_layer_kernel(const float* __restrict__ W,
                                  const float* __restrict__ g, const float* __restrict__ bb,
                                  const float* __restrict__ m, const float* __restrict__ v,
                                  _Float16* __restrict__ Wh, float* __restrict__ ss,
                                  int N, int K, int Kpad) {
  int t = blockIdx.x * blockDim.x + threadIdx.x;
  int total = N * Kpad;
  if (t < total) {
    int n = t / Kpad, k = t % Kpad;
    Wh[t] = (k < K) ? (_Float16)W[(size_t)n * K + k] : (_Float16)0.f;
  }
  if (t < N) {
    float s = g[t] * rsqrtf(v[t] + EPS_BN);
    ss[t] = s;
    ss[N + t] = bb[t] - m[t] * s;
  }
}

// ---------------------------------------------------------------------------
// WMMA GEMM + fused BN + ReLU:  Y[m,n] = relu((X @ Wh^T)[m,n]*scale[n]+shift[n])
// X: [M, Kpad] f16, Wh: [N, Kpad] f16 (both zero-padded past real K).
// Block = 8 waves -> 128(M) x 32(N) tile; each wave: 32x16 (2 WMMAs/K-step).
// Double-buffered LDS tiles; async global->LDS staging when available,
// one barrier per K-step (preceded by s_wait_asynccnt on the async path).
// ---------------------------------------------------------------------------
__global__ __launch_bounds__(256) void gemm_bn_relu_kernel(
    const _Float16* __restrict__ X, const _Float16* __restrict__ Wh,
    const float* __restrict__ ss, _Float16* __restrict__ Y,
    int M, int N, int Kpad) {
  __shared__ _Float16 Xs[2][128][32];
  __shared__ _Float16 Ws[2][32][32];
  const int tid  = threadIdx.x;
  const int lane = tid & 31;
  const int wave = tid >> 5;
  const int wm = wave & 3;   // 4 M-subtiles of 32 rows
  const int wn = wave >> 2;  // 2 N-subtiles of 16 cols
  const int tileM = blockIdx.y * 128;
  const int tileN = blockIdx.x * 32;
  const int xr = tid >> 1;          // 0..127
  const int xc = (tid & 1) * 16;    // 0 or 16
  const int wr = tid >> 3;          // 0..31
  const int wc = (tid & 7) * 4;     // 0..28
  const _Float16* Xrow = X + (size_t)(tileM + xr) * Kpad + xc;
  const _Float16* Wrow = Wh + (size_t)(tileN + wr) * Kpad + wc;

  v8f acc0 = {};
  v8f acc1 = {};

  // prologue: tile 0 into buffer 0
#if ASYNC_LDS
  async_b128(Xrow,     &Xs[0][xr][xc]);
  async_b128(Xrow + 8, &Xs[0][xr][xc + 8]);
  async_b64 (Wrow,     &Ws[0][wr][wc]);
  wait_async();
#else
  *(v8h*)&Xs[0][xr][xc]     = *(const v8h*)(Xrow);
  *(v8h*)&Xs[0][xr][xc + 8] = *(const v8h*)(Xrow + 8);
  *(v4h*)&Ws[0][wr][wc]     = *(const v4h*)(Wrow);
#endif
  __syncthreads();

  const int nk = Kpad >> 5;
  const int half = lane >> 4;
  const int mrow0 = wm * 32 + (lane & 15);
  const int ncol  = wn * 16 + (lane & 15);
  for (int t0 = 0; t0 < nk; ++t0) {
    const int buf = t0 & 1;
    if (t0 + 1 < nk) {
      const int k1 = (t0 + 1) << 5;
#if ASYNC_LDS
      async_b128(Xrow + k1,     &Xs[buf ^ 1][xr][xc]);
      async_b128(Xrow + k1 + 8, &Xs[buf ^ 1][xr][xc + 8]);
      async_b64 (Wrow + k1,     &Ws[buf ^ 1][wr][wc]);
#else
      *(v8h*)&Xs[buf ^ 1][xr][xc]     = *(const v8h*)(Xrow + k1);
      *(v8h*)&Xs[buf ^ 1][xr][xc + 8] = *(const v8h*)(Xrow + k1 + 8);
      *(v4h*)&Ws[buf ^ 1][wr][wc]     = *(const v4h*)(Wrow + k1);
#endif
    }
    union VA { v16h v; v8h h[2]; } a0, a1, bf;
    a0.h[0] = *(const v8h*)&Xs[buf][mrow0][8 * half];
    a0.h[1] = *(const v8h*)&Xs[buf][mrow0][16 + 8 * half];
    a1.h[0] = *(const v8h*)&Xs[buf][mrow0 + 16][8 * half];
    a1.h[1] = *(const v8h*)&Xs[buf][mrow0 + 16][16 + 8 * half];
    bf.h[0] = *(const v8h*)&Ws[buf][ncol][8 * half];
    bf.h[1] = *(const v8h*)&Ws[buf][ncol][16 + 8 * half];
    acc0 = __builtin_amdgcn_wmma_f32_16x16x32_f16(false, a0.v, false, bf.v,
                                                  (short)0, acc0, false, false);
    acc1 = __builtin_amdgcn_wmma_f32_16x16x32_f16(false, a1.v, false, bf.v,
                                                  (short)0, acc1, false, false);
#if ASYNC_LDS
    wait_async();  // tile t+1 fully in LDS before the publishing barrier
#endif
    __syncthreads();
  }

  const int nG = tileN + ncol;
  const float s  = ss[nG];
  const float sh = ss[N + nG];
  const int mBase = tileM + wm * 32 + half * 8;
#pragma unroll
  for (int r = 0; r < 8; ++r) {
    float y0 = fmaxf(acc0[r] * s + sh, 0.0f);
    float y1 = fmaxf(acc1[r] * s + sh, 0.0f);
    Y[(size_t)(mBase + r) * N + nG]      = (_Float16)y0;
    Y[(size_t)(mBase + 16 + r) * N + nG] = (_Float16)y1;
  }
}

// ---------------------------------------------------------------------------
// Max-pool over neighborhood dim: in [P, ns, C] -> out [P, C]
// ---------------------------------------------------------------------------
__global__ void maxpool_h_kernel(const _Float16* __restrict__ in, _Float16* __restrict__ out,
                                 int ns, int C, int total) {
  int t = blockIdx.x * blockDim.x + threadIdx.x;
  if (t >= total) return;
  int p = t / C, c = t % C;
  const _Float16* src = in + (size_t)p * ns * C + c;
  float mx = -1e30f;
  for (int j = 0; j < ns; ++j) mx = fmaxf(mx, (float)src[(size_t)j * C]);
  out[(size_t)p * C + c] = (_Float16)mx;
}

__global__ void maxpool_f_kernel(const _Float16* __restrict__ in, float* __restrict__ out,
                                 int ns, int C, int total) {
  int t = blockIdx.x * blockDim.x + threadIdx.x;
  if (t >= total) return;
  int p = t / C, c = t % C;
  const _Float16* src = in + (size_t)p * ns * C + c;
  float mx = -1e30f;
  for (int j = 0; j < ns; ++j) mx = fmaxf(mx, (float)src[(size_t)j * C]);
  out[(size_t)p * C + c] = mx;
}

// ---------------------------------------------------------------------------
// Host orchestration
// ---------------------------------------------------------------------------
extern "C" void kernel_launch(void* const* d_in, const int* in_sizes, int n_in,
                              void* d_out, int out_size, void* d_ws, size_t ws_size,
                              hipStream_t stream) {
  (void)in_sizes; (void)n_in; (void)out_size; (void)ws_size;
  const float* pc = (const float*)d_in[0];
  auto lp = [&](int s, int l, int q) -> const float* {
    return (const float*)d_in[1 + (s * 3 + l) * 5 + q];  // w,g,b,m,v
  };

  // Workspace arena
  char* wsb = (char*)d_ws;
  size_t off = 0;
  auto alloc = [&](size_t bytes) -> void* {
    void* p = wsb + off;
    off = (off + bytes + 255) & ~(size_t)255;
    return p;
  };
  const int B = 8, N0 = 8192;
  int*      idx1  = (int*)alloc((size_t)B * 512 * 4);
  float*    nx1   = (float*)alloc((size_t)B * 512 * 3 * 4);
  int*      nidx1 = (int*)alloc((size_t)B * 512 * 64 * 4);
  _Float16* f1    = (_Float16*)alloc((size_t)B * 512 * 64 * 2);
  int*      idx2  = (int*)alloc((size_t)B * 128 * 4);
  float*    nx2   = (float*)alloc((size_t)B * 128 * 3 * 4);
  int*      nidx2 = (int*)alloc((size_t)B * 128 * 32 * 4);
  _Float16* f2    = (_Float16*)alloc((size_t)B * 128 * 256 * 2);
  int*      idx3  = (int*)alloc((size_t)B * 32 * 4);
  float*    nx3   = (float*)alloc((size_t)B * 32 * 3 * 4);
  int*      nidx3 = (int*)alloc((size_t)B * 32 * 16 * 4);
  _Float16* f3    = (_Float16*)alloc((size_t)B * 32 * 512 * 2);
  _Float16* Whbuf = (_Float16*)alloc((size_t)1024 * 1024 * 2);  // 2 MB weights (f16)
  float*    ssbuf = (float*)alloc((size_t)2 * 1024 * 4);        // scale/shift
  _Float16* bufA  = (_Float16*)alloc((size_t)262144 * 32 * 2);  // 16 MB ping
  _Float16* bufB  = (_Float16*)alloc((size_t)262144 * 64 * 2);  // 32 MB pong

  auto run_layer = [&](const _Float16* X, _Float16* Y, int M, int Nc, int K, int Kpad,
                       int s, int l) {
    int total = Nc * Kpad;
    prep_layer_kernel<<<(total + 255) / 256, 256, 0, stream>>>(
        lp(s, l, 0), lp(s, l, 1), lp(s, l, 2), lp(s, l, 3), lp(s, l, 4),
        Whbuf, ssbuf, Nc, K, Kpad);
    dim3 grid(Nc / 32, M / 128);
    gemm_bn_relu_kernel<<<grid, 256, 0, stream>>>(X, Whbuf, ssbuf, Y, M, Nc, Kpad);
  };

  // ---------------- Stage 1: N=8192 -> np=512, ns=64, r=0.04, 3->32->32->64
  fps_kernel<<<B, 256, 0, stream>>>(pc, idx1, N0, 512);
  gather_xyz_kernel<<<(B * 512 + 255) / 256, 256, 0, stream>>>(pc, idx1, nx1, N0, 512, B * 512);
  ballquery_kernel<<<(B * 512 * 32 + 255) / 256, 256, 0, stream>>>(
      pc, nx1, nidx1, B, N0, 512, 64, 0.04f * 0.04f);
  {
    int total = B * 512 * 64;
    group_kernel<<<(total + 255) / 256, 256, 0, stream>>>(
        pc, nx1, (const _Float16*)nullptr, nidx1, bufA, N0, 512, 64, 0, 32, total);
  }
  run_layer(bufA, bufB, B * 512 * 64, 32, 3, 32, 0, 0);
  run_layer(bufB, bufA, B * 512 * 64, 32, 32, 32, 0, 1);
  run_layer(bufA, bufB, B * 512 * 64, 64, 32, 32, 0, 2);
  maxpool_h_kernel<<<(B * 512 * 64 + 255) / 256, 256, 0, stream>>>(bufB, f1, 64, 64, B * 512 * 64);

  // ---------------- Stage 2: N=512 -> np=128, ns=32, r=0.08, 67->128->128->256
  fps_kernel<<<B, 256, 0, stream>>>(nx1, idx2, 512, 128);
  gather_xyz_kernel<<<(B * 128 + 255) / 256, 256, 0, stream>>>(nx1, idx2, nx2, 512, 128, B * 128);
  ballquery_kernel<<<(B * 128 * 32 + 255) / 256, 256, 0, stream>>>(
      nx1, nx2, nidx2, B, 512, 128, 32, 0.08f * 0.08f);
  {
    int total = B * 128 * 32;
    group_kernel<<<(total + 255) / 256, 256, 0, stream>>>(
        nx1, nx2, f1, nidx2, bufA, 512, 128, 32, 64, 96, total);
  }
  run_layer(bufA, bufB, B * 128 * 32, 128, 67, 96, 1, 0);
  run_layer(bufB, bufA, B * 128 * 32, 128, 128, 128, 1, 1);
  run_layer(bufA, bufB, B * 128 * 32, 256, 128, 128, 1, 2);
  maxpool_h_kernel<<<(B * 128 * 256 + 255) / 256, 256, 0, stream>>>(bufB, f2, 32, 256, B * 128 * 256);

  // ---------------- Stage 3: N=128 -> np=32, ns=16, r=0.16, 259->256->512->512
  fps_kernel<<<B, 256, 0, stream>>>(nx2, idx3, 128, 32);
  gather_xyz_kernel<<<(B * 32 + 255) / 256, 256, 0, stream>>>(nx2, idx3, nx3, 128, 32, B * 32);
  ballquery_kernel<<<(B * 32 * 32 + 255) / 256, 256, 0, stream>>>(
      nx2, nx3, nidx3, B, 128, 32, 16, 0.16f * 0.16f);
  {
    int total = B * 32 * 16;
    group_kernel<<<(total + 255) / 256, 256, 0, stream>>>(
        nx2, nx3, f2, nidx3, bufA, 128, 32, 16, 256, 288, total);
  }
  run_layer(bufA, bufB, B * 32 * 16, 256, 259, 288, 2, 0);
  run_layer(bufB, bufA, B * 32 * 16, 512, 256, 256, 2, 1);
  run_layer(bufA, bufB, B * 32 * 16, 512, 512, 512, 2, 2);
  maxpool_h_kernel<<<(B * 32 * 512 + 255) / 256, 256, 0, stream>>>(bufB, f3, 16, 512, B * 32 * 512);

  // ---------------- Stage 4: global MLP over 32 points, 515->512->1024->1024
  pack4_kernel<<<1, 256, 0, stream>>>(nx3, f3, bufA, 512, 544, 256);
  run_layer(bufA, bufB, 256, 512, 515, 544, 3, 0);
  run_layer(bufB, bufA, 256, 1024, 512, 512, 3, 1);
  run_layer(bufA, bufB, 256, 1024, 1024, 1024, 3, 2);
  maxpool_f_kernel<<<(B * 1024 + 255) / 256, 256, 0, stream>>>(bufB, (float*)d_out, 32, 1024, B * 1024);
}
